// STSGCN_52707838656913
// MI455X (gfx1250) — matrix operand order repaired and checked by reading, not compiled, not run on hardware
//
#include <hip/hip_runtime.h>
#include <hip/hip_bf16.h>
#include <math.h>

// ---------------------------------------------------------------------------
// Types for CDNA5 WMMA (wave32): bf16 16x16x32, f32 accumulate.
// ---------------------------------------------------------------------------
typedef __attribute__((ext_vector_type(16))) __bf16 v16bf;
typedef __attribute__((ext_vector_type(8)))  __bf16 v8bf;
typedef __attribute__((ext_vector_type(8)))  float  v8f;

#define N_NODES   1024
#define N3        3072
#define HCH       64
#define WCOLS     512      // columns per window = B * H = 8 * 64
#define BATCH     8

// ---------------------------------------------------------------------------
// adj = bf16(mask * A)
// ---------------------------------------------------------------------------
__global__ void k_prep_adj(const float* __restrict__ A, const float* __restrict__ M,
                           __bf16* __restrict__ adjb, int n) {
    int i = blockIdx.x * 256 + threadIdx.x;
    if (i < n) adjb[i] = (__bf16)(A[i] * M[i]);
}

__global__ void k_cvt_bf16(const float* __restrict__ s, __bf16* __restrict__ d, int n) {
    int i = blockIdx.x * 256 + threadIdx.x;
    if (i < n) d[i] = (__bf16)s[i];
}

// ---------------------------------------------------------------------------
// Input 1x1 conv + relu: (B,12,N,3) -> (B,12,N,64)
// ---------------------------------------------------------------------------
__global__ void k_in_conv(const float* __restrict__ x, const float* __restrict__ w,
                          const float* __restrict__ b, float* __restrict__ out, int n) {
    int idx = blockIdx.x * 256 + threadIdx.x;
    if (idx >= n) return;
    int h = idx & 63;
    int s = idx >> 6;                       // (b*12 + t)*1024 + node
    const float* xp = x + (size_t)s * 3;
    float v = b[h] + xp[0] * w[h*3+0] + xp[1] * w[h*3+1] + xp[2] * w[h*3+2];
    out[idx] = fmaxf(v, 0.f);
}

// ---------------------------------------------------------------------------
// Pack a group of windows [winBase, winBase+nWin) into Ht (k-major bf16),
// adding temb+semb.  Global column colg = win*512 + b*64 + c.
// Ht[colg*3072 + r], r = toff*1024 + node.
// ---------------------------------------------------------------------------
__global__ void k_pack_win(const float* __restrict__ din, const float* __restrict__ temb,
                           const float* __restrict__ semb, __bf16* __restrict__ Ht,
                           int Tl, int winBase, int nTot) {
    int tid = blockIdx.x * 256 + threadIdx.x;
    if (tid >= nTot) return;
    int colg = tid / N3;
    int r    = tid - colg * N3;
    int win  = colg >> 9;
    int col  = colg & 511;
    int b    = col >> 6, c = col & 63;
    int toff = r >> 10, node = r & 1023;
    int tt   = winBase + win + toff;
    float v = din[(((size_t)b * Tl + tt) * N_NODES + node) * HCH + c]
            + temb[tt * HCH + c]
            + semb[node * HCH + c];
    Ht[tid] = (__bf16)v;
}

// ---------------------------------------------------------------------------
// GEMM1: hv(3072, ldN) = adjb(3072,3072) x H(3072, ldN), H k-major (Ht).
// ldN = nWin*512.  Output stored bf16 row-major (feeds GEMM2's A operand).
// Block = 256 threads (8 waves). Block tile 128x128; wave tile 64x32.
// ---------------------------------------------------------------------------
__global__ __launch_bounds__(256) void k_gemm1(const __bf16* __restrict__ adjb,
                                               const __bf16* __restrict__ Ht,
                                               __bf16* __restrict__ hv, int ldN) {
    const int lane  = threadIdx.x & 31;
    const int wave  = threadIdx.x >> 5;
    const int waveM = wave >> 2;            // 0..1
    const int waveN = wave & 3;             // 0..3
    const int row0  = blockIdx.x * 128 + waveM * 64;
    const int col0  = blockIdx.y * 128 + waveN * 32;
    const int lr    = lane & 15;            // M (A) / N (B) index within tile
    const int hl    = lane >> 4;            // lane half

    v8f acc[4][2];
#pragma unroll
    for (int mt = 0; mt < 4; ++mt)
#pragma unroll
        for (int nt = 0; nt < 2; ++nt)
            acc[mt][nt] = (v8f){0.f,0.f,0.f,0.f,0.f,0.f,0.f,0.f};

    for (int k0 = 0; k0 < N3; k0 += 32) {
        v16bf a[4];
#pragma unroll
        for (int mt = 0; mt < 4; ++mt) {
            // ISA 16-bit A layout: lane lr holds row M; halves split K in 8-chunks.
            const __bf16* p = adjb + (size_t)(row0 + mt*16 + lr) * N3 + k0 + hl*8;
            v8bf lo = *(const v8bf*)p;
            v8bf hi = *(const v8bf*)(p + 16);
            a[mt] = __builtin_shufflevector(lo, hi, 0,1,2,3,4,5,6,7,8,9,10,11,12,13,14,15);
            __builtin_prefetch(p + 32, 0, 1);   // unconditional: branch-free loop
        }
        v16bf bb[2];
#pragma unroll
        for (int nt = 0; nt < 2; ++nt) {
            // Ht is k-major: two 16B chunks per lane, K split in 8-chunks like A.
            const __bf16* p = Ht + (size_t)(col0 + nt*16 + lr) * N3 + k0 + hl*8;
            v8bf blo = *(const v8bf*)p;
            v8bf bhi = *(const v8bf*)(p + 16);
            bb[nt] = __builtin_shufflevector(blo, bhi, 0,1,2,3,4,5,6,7,8,9,10,11,12,13,14,15);
        }
#pragma unroll
        for (int mt = 0; mt < 4; ++mt)
#pragma unroll
            for (int nt = 0; nt < 2; ++nt)
                acc[mt][nt] = __builtin_amdgcn_wmma_f32_16x16x32_bf16(
                    false, a[mt], false, bb[nt], (short)0, acc[mt][nt], false, false);
    }
#pragma unroll
    for (int mt = 0; mt < 4; ++mt)
#pragma unroll
        for (int nt = 0; nt < 2; ++nt) {
            int n = col0 + nt*16 + lr;
#pragma unroll
            for (int j = 0; j < 8; ++j) {
                int m = row0 + mt*16 + hl*8 + j;      // C/D layout: VGPR j -> M = hl*8+j
                hv[(size_t)m * ldN + n] = (__bf16)acc[mt][nt][j];
            }
        }
}

// ---------------------------------------------------------------------------
// GEMM2 + bias + GLU, batched over windows:
//   y(3072,128) = hv[win,b](3072,64) x W[win,step]^T
// Writes next h into Ht (k-major bf16, in place) and max-reduces the middle
// slice (rows 1024..2047) into the next layer's activation buffer.
// Block = 256 threads (8 waves); wave tile 16x128; grid (24, B, nWin).
// ---------------------------------------------------------------------------
__global__ __launch_bounds__(256) void k_gemm2_glu(const __bf16* __restrict__ hv, int ldN,
                                                   const __bf16* __restrict__ Wlayer,
                                                   const float* __restrict__ blayer,
                                                   __bf16* __restrict__ Ht,
                                                   float* __restrict__ dOut,
                                                   int TlOut, int winBase, int step) {
    const int lane = threadIdx.x & 31;
    const int wave = threadIdx.x >> 5;
    const int m0   = blockIdx.x * 128 + wave * 16;
    const int b    = blockIdx.y;
    const int win  = blockIdx.z;
    const int iwin = winBase + win;
    const int bcolg = win * WCOLS + b * HCH;
    const int lr   = lane & 15;
    const int hl   = lane >> 4;

    const __bf16* Wb   = Wlayer + (size_t)(iwin * 3 + step) * 128 * 64;
    const float*  bias = blayer + (size_t)(iwin * 3 + step) * 128;

    v8f acc[8];
#pragma unroll
    for (int nt = 0; nt < 8; ++nt)
        acc[nt] = (v8f){0.f,0.f,0.f,0.f,0.f,0.f,0.f,0.f};

#pragma unroll
    for (int k0 = 0; k0 < 64; k0 += 32) {
        // A fragment straight from bf16 hv (row-major, c contiguous).
        const __bf16* ap = hv + (size_t)(m0 + lr) * ldN + bcolg + k0 + hl*8;
        v8bf lo = *(const v8bf*)ap;
        v8bf hi = *(const v8bf*)(ap + 16);
        v16bf af = __builtin_shufflevector(lo, hi, 0,1,2,3,4,5,6,7,8,9,10,11,12,13,14,15);
#pragma unroll
        for (int nt = 0; nt < 8; ++nt) {
            // B[k=c][n=o] = W[o,c]; W row-major is exactly n-major/k-contiguous.
            const __bf16* bp = Wb + (size_t)(nt*16 + lr) * 64 + k0 + hl*8;
            v8bf blo = *(const v8bf*)bp;
            v8bf bhi = *(const v8bf*)(bp + 16);
            v16bf bf = __builtin_shufflevector(blo, bhi, 0,1,2,3,4,5,6,7,8,9,10,11,12,13,14,15);
            acc[nt] = __builtin_amdgcn_wmma_f32_16x16x32_bf16(
                false, af, false, bf, (short)0, acc[nt], false, false);
        }
    }

#pragma unroll
    for (int nt = 0; nt < 4; ++nt) {
        int   c  = nt*16 + lr;          // output channel (GLU lhs); rhs is c+64
        float bl = bias[c];
        float br = bias[c + 64];
#pragma unroll
        for (int j = 0; j < 8; ++j) {
            float lhs = acc[nt][j]     + bl;
            float rhs = acc[nt + 4][j] + br;
            float g   = lhs * (1.f / (1.f + __expf(-rhs)));
            int   m   = m0 + hl*8 + j;
            Ht[(size_t)(bcolg + c) * N3 + m] = (__bf16)g;   // next h, k-major
            if (m >= N_NODES && m < 2 * N_NODES) {          // middle time slice
                float* dst = dOut + ((((size_t)b * TlOut + iwin) * N_NODES) + (m - N_NODES)) * HCH + c;
                *dst = (step == 0) ? g : fmaxf(*dst, g);
            }
        }
    }
}

// ---------------------------------------------------------------------------
// Output heads: dF (B,4,N,64) -> out (B,12,N). Thread per (b,o,n).
// ---------------------------------------------------------------------------
__global__ __launch_bounds__(64) void k_heads(const float* __restrict__ dF,
                                              const float* __restrict__ w1,
                                              const float* __restrict__ b1,
                                              const float* __restrict__ w2,
                                              const float* __restrict__ b2,
                                              float* __restrict__ out) {
    const int n = blockIdx.x * 64 + threadIdx.x;
    const int o = blockIdx.y;
    const int b = blockIdx.z;

    const float* xp0 = dF + (((size_t)b * 4 + 0) * N_NODES + n) * HCH;
    const float* xp1 = dF + (((size_t)b * 4 + 1) * N_NODES + n) * HCH;
    const float* xp2 = dF + (((size_t)b * 4 + 2) * N_NODES + n) * HCH;
    const float* xp3 = dF + (((size_t)b * 4 + 3) * N_NODES + n) * HCH;

    const float* W1o = w1 + (size_t)o * 128 * 256;
    const float* W2o = w2 + (size_t)o * 128;
    const float* B1o = b1 + (size_t)o * 128;

    float s2 = b2[o];
    for (int k = 0; k < 128; ++k) {
        const float* wr = W1o + (size_t)k * 256;
        float s = B1o[k];
#pragma unroll 8
        for (int c = 0; c < 64; ++c) {
            s += xp0[c] * wr[c]       + xp1[c] * wr[64 + c]
               + xp2[c] * wr[128 + c] + xp3[c] * wr[192 + c];
        }
        s2 += fmaxf(s, 0.f) * W2o[k];
    }
    out[((size_t)b * 12 + o) * N_NODES + n] = s2;
}

// ---------------------------------------------------------------------------
// Host orchestration (graph-capture safe: stream launches only).
// ---------------------------------------------------------------------------
extern "C" void kernel_launch(void* const* d_in, const int* in_sizes, int n_in,
                              void* d_out, int out_size, void* d_ws, size_t ws_size,
                              hipStream_t stream) {
    const float* x    = (const float*)d_in[0];
    const float* A    = (const float*)d_in[1];
    const float* mask = (const float*)d_in[2];
    const float* in_w = (const float*)d_in[3];
    const float* in_b = (const float*)d_in[4];
    const float* gcn_w[4] = {(const float*)d_in[5],  (const float*)d_in[9],
                             (const float*)d_in[13], (const float*)d_in[17]};
    const float* gcn_b[4] = {(const float*)d_in[6],  (const float*)d_in[10],
                             (const float*)d_in[14], (const float*)d_in[18]};
    const float* temb[4]  = {(const float*)d_in[7],  (const float*)d_in[11],
                             (const float*)d_in[15], (const float*)d_in[19]};
    const float* semb[4]  = {(const float*)d_in[8],  (const float*)d_in[12],
                             (const float*)d_in[16], (const float*)d_in[20]};
    const float* ow1 = (const float*)d_in[21];
    const float* ob1 = (const float*)d_in[22];
    const float* ow2 = (const float*)d_in[23];
    const float* ob2 = (const float*)d_in[24];

    // Fixed workspace layout (bytes, 256-aligned):
    //   adjb : 3072*3072 bf16   = 18,874,368
    //   Wb   : 688,128 bf16     =  1,376,256
    //   dA   : 8*12*1024*64 f32 = 25,165,824
    //   dB   : 8*10*1024*64 f32 = 20,971,520      fixed total = 66,387,968
    // Then per batched window: Ht 3,145,728 + hv 3,145,728 = 6,291,456 bytes.
    char* ws = (char*)d_ws;
    __bf16* adjb = (__bf16*)(ws);
    __bf16* Wb   = (__bf16*)(ws + 18874368);
    float*  dA   = (float*) (ws + 20250624);
    float*  dB   = (float*) (ws + 45416448);
    const size_t fixedEnd = 66387968;

    // Window-group size from available scratch (deterministic: ws_size fixed).
    int G = 1;
    if (ws_size > fixedEnd) {
        size_t g = (ws_size - fixedEnd) / 6291456u;
        G = (g < 1) ? 1 : (g > 10 ? 10 : (int)g);
    }
    __bf16* Ht = (__bf16*)(ws + fixedEnd);
    __bf16* hv = (__bf16*)(ws + fixedEnd + (size_t)G * 3145728u);

    // 1) adjacency -> bf16 (stays hot in 192MB L2 across all layer GEMMs)
    k_prep_adj<<<(N3 * N3 + 255) / 256, 256, 0, stream>>>(A, mask, adjb, N3 * N3);

    // 2) GCN weights -> bf16 (concatenated)
    static const int    wcnt[4]   = {245760, 196608, 147456, 98304};
    static const size_t layoff[4] = {0, 245760, 442368, 589824};
    for (int l = 0; l < 4; ++l)
        k_cvt_bf16<<<(wcnt[l] + 255) / 256, 256, 0, stream>>>(gcn_w[l], Wb + layoff[l], wcnt[l]);

    // 3) input conv
    const int nin = BATCH * 12 * N_NODES * HCH;
    k_in_conv<<<(nin + 255) / 256, 256, 0, stream>>>(x, in_w, in_b, dA, nin);

    // 4) four STSGCL layers, windows batched in groups of up to G
    static const int TL[4] = {12, 10, 8, 6};
    float* bufs[2] = {dA, dB};
    int cur = 0;
    for (int l = 0; l < 4; ++l) {
        const int Tl = TL[l];
        const int nWinTot = Tl - 2;
        float* din  = bufs[cur];
        float* dout = bufs[1 - cur];
        for (int done = 0; done < nWinTot; ) {
            const int g   = (nWinTot - done < G) ? (nWinTot - done) : G;
            const int ldN = g * WCOLS;
            k_pack_win<<<(g * WCOLS * (N3 / 256)), 256, 0, stream>>>(
                din, temb[l], semb[l], Ht, Tl, done, g * WCOLS * N3);
            for (int step = 0; step < 3; ++step) {
                k_gemm1<<<dim3(N3 / 128, ldN / 128), 256, 0, stream>>>(adjb, Ht, hv, ldN);
                k_gemm2_glu<<<dim3(N3 / 128, BATCH, g), 256, 0, stream>>>(
                    hv, ldN, Wb + layoff[l], gcn_b[l], Ht, dout, nWinTot, done, step);
            }
            done += g;
        }
        cur ^= 1;
    }

    // 5) output heads (final activations in bufs[cur] == dA, shape (B,4,N,64))
    k_heads<<<dim3(N_NODES / 64, 12, BATCH), 64, 0, stream>>>(
        bufs[cur], ow1, ob1, ow2, ob2, (float*)d_out);
}
